// RelationalReasonerV2_51934744543297
// MI455X (gfx1250) — matrix operand order repaired and checked by reading, not compiled, not run on hardware
//
#include <hip/hip_runtime.h>
#include <hip/hip_bf16.h>

// Problem constants (from reference)
#define Bz   4
#define Nn   160
#define Dd   64
#define ZIc  64
#define ZEc  64
#define Ll   4
#define LEc  8
#define H1c  128
#define H2c  64

#define PAIRS      (Nn * Nn)          // 25600
#define GROUPS_PER (PAIRS / 16)       // 1600 pair-groups of 16

typedef __attribute__((ext_vector_type(16))) _Float16 v16h;
typedef __attribute__((ext_vector_type(8)))  float    v8f;

// ---------------------------------------------------------------------------
// Prep 1: row features  pi = ne@Wi1[0:64], pj = ne@Wi1[64:128],
//                        qi = ne@Wl1[0:64], qj = ne@Wl1[64:128]
// Each output matrix is (B*N, H1) = (640,128) fp32.
// ---------------------------------------------------------------------------
__global__ __launch_bounds__(256) void prep_rows_kernel(
    const float* __restrict__ ne,   // (B,N,D)
    const float* __restrict__ Wi1,  // (192,128)
    const float* __restrict__ Wl1,  // (200,128)
    float* __restrict__ pi, float* __restrict__ pj,
    float* __restrict__ qi, float* __restrict__ qj)
{
    const int matsz = Bz * Nn * H1c;                 // 81920
    int gid = blockIdx.x * 256 + threadIdx.x;        // < 4*matsz
    int mat = gid / matsz;
    int r   = gid % matsz;
    int bn  = r / H1c;                               // 0..639
    int h   = r % H1c;

    const float* nerow = ne + bn * Dd;
    const float* Wrow;
    float* dst;
    switch (mat) {
        case 0:  Wrow = Wi1;             dst = pi; break;
        case 1:  Wrow = Wi1 + Dd * H1c;  dst = pj; break;
        case 2:  Wrow = Wl1;             dst = qi; break;
        default: Wrow = Wl1 + Dd * H1c;  dst = qj; break;
    }
    float acc = 0.0f;
#pragma unroll 8
    for (int d = 0; d < Dd; ++d)
        acc += nerow[d] * Wrow[d * H1c + h];
    dst[r] = acc;
}

// ---------------------------------------------------------------------------
// Prep 2: pzb[b,h]   = bi1[h] + z_intra[b]@Wi1[128:192]
//         qzl[b,l,h] = bl1[h] + z_inter[b]@Wl1[128:192] + lag[l]@Wl1[192:200]
//         Wi2h/Wl2h  = f16 TRANSPOSED (n-major, (64,128)) layer-2 weights so
//                      each WMMA B-fragment lane reads 16 contiguous halves.
// total threads = 512 + 2048 + 8192 + 8192 = 18944 = 74 * 256 exactly.
// ---------------------------------------------------------------------------
__global__ __launch_bounds__(256) void prep_misc_kernel(
    const float* __restrict__ z_intra, const float* __restrict__ z_inter,
    const float* __restrict__ lag,
    const float* __restrict__ Wi1, const float* __restrict__ bi1,
    const float* __restrict__ Wl1, const float* __restrict__ bl1,
    const float* __restrict__ Wi2, const float* __restrict__ Wl2,
    float* __restrict__ pzb, float* __restrict__ qzl,
    _Float16* __restrict__ Wi2hT, _Float16* __restrict__ Wl2hT)
{
    int gid = blockIdx.x * 256 + threadIdx.x;
    if (gid < Bz * H1c) {                                     // 512: pzb
        int b = gid / H1c, h = gid % H1c;
        float a = bi1[h];
#pragma unroll 8
        for (int z = 0; z < ZIc; ++z)
            a += z_intra[b * ZIc + z] * Wi1[(2 * Dd + z) * H1c + h];
        pzb[gid] = a;
    } else if (gid < Bz * H1c + Bz * Ll * H1c) {              // 2048: qzl
        int r = gid - Bz * H1c;
        int bl = r / H1c, h = r % H1c;
        int b = bl / Ll, l = bl % Ll;
        float a = bl1[h];
#pragma unroll 8
        for (int z = 0; z < ZEc; ++z)
            a += z_inter[b * ZEc + z] * Wl1[(2 * Dd + z) * H1c + h];
#pragma unroll
        for (int e = 0; e < LEc; ++e)
            a += lag[l * LEc + e] * Wl1[(2 * Dd + ZEc + e) * H1c + h];
        qzl[r] = a;
    } else if (gid < Bz * H1c + Bz * Ll * H1c + H1c * H2c) {  // 8192: Wi2hT
        int r = gid - (Bz * H1c + Bz * Ll * H1c);
        int n = r / H1c, k = r % H1c;
        Wi2hT[r] = (_Float16)Wi2[k * H2c + n];                // transpose
    } else {                                                  // 8192: Wl2hT
        int r = gid - (Bz * H1c + Bz * Ll * H1c + H1c * H2c);
        int n = r / H1c, k = r % H1c;
        Wl2hT[r] = (_Float16)Wl2[k * H2c + n];                // transpose
    }
}

// Load 8 consecutive h1 values (k0..k0+7, k0 % 4 == 0) from the three fp32
// sources as float4 pairs, relu, and pack into afrag elements e0..e0+7.
__device__ __forceinline__ void fill8(v16h& frag, int e0,
                                      const float4* __restrict__ a4,
                                      const float4* __restrict__ b4,
                                      const float4* __restrict__ z4,
                                      int k0)
{
    const int q = k0 >> 2;
    float4 av0 = a4[q], av1 = a4[q + 1];
    float4 bv0 = b4[q], bv1 = b4[q + 1];
    float4 zv0 = z4[q], zv1 = z4[q + 1];
    float vals[8] = {
        av0.x + bv0.x + zv0.x, av0.y + bv0.y + zv0.y,
        av0.z + bv0.z + zv0.z, av0.w + bv0.w + zv0.w,
        av1.x + bv1.x + zv1.x, av1.y + bv1.y + zv1.y,
        av1.z + bv1.z + zv1.z, av1.w + bv1.w + zv1.w };
#pragma unroll
    for (int e = 0; e < 8; ++e) {
        float v = vals[e] > 0.0f ? vals[e] : 0.0f;
        frag[e0 + e] = (_Float16)v;
    }
}

// ---------------------------------------------------------------------------
// Main fused pair-MLP kernel. One wave32 = 16 pairs (one WMMA M-tile).
//   h1 = relu(rowi[i] + rowj[j] + zbase)          (128, built as f16 A-frags)
//   h2 = relu(h1 @ W2 + b2)                        (64, via 16x16x32 f16 WMMA)
//   logit = h2 @ w3 + b3 ; out = sigmoid * (i!=j)
// ---------------------------------------------------------------------------
__global__ __launch_bounds__(256) void pair_mlp_wmma_kernel(
    const float*   __restrict__ rowi,   // (B,N,128)
    const float*   __restrict__ rowj,   // (B,N,128)
    const float*   __restrict__ zbase,  // (B*Lg,128), includes layer-1 bias
    const _Float16* __restrict__ WhT,   // (64,128) f16, n-major (transposed)
    const float*   __restrict__ b2,     // (64)
    const float*   __restrict__ w3,     // (64)
    const float*   __restrict__ b3,     // (1)
    float*         __restrict__ out,    // (B*Lg, N*N)
    int Lg)
{
    const int wave = threadIdx.x >> 5;
    const int lane = threadIdx.x & 31;
    const int Gid  = blockIdx.x * 8 + wave;
    const int zIdx = Gid / GROUPS_PER;          // b (W) or b*L+l (A)
    const int g    = Gid % GROUPS_PER;
    const int b    = zIdx / Lg;
    const int p0   = g * 16;

    const int m    = lane & 15;                 // row within the 16-pair tile
    const int half = lane >> 4;                 // 0: lanes 0-15, 1: lanes 16-31
    const int p    = p0 + m;
    const int i    = p / Nn;
    const int j    = p % Nn;

    const float4* ri4 = reinterpret_cast<const float4*>(rowi  + (b * Nn + i) * H1c);
    const float4* rj4 = reinterpret_cast<const float4*>(rowj  + (b * Nn + j) * H1c);
    const float4* zb4 = reinterpret_cast<const float4*>(zbase + zIdx * H1c);

    // ---- build A fragments: h1 = relu(ri + rj + zb), f16, 4 K-steps of 32.
    // Contiguous k order used here (B matches): lanes 0-15 hold K = base+{0..15}; lanes 16-31
    // hold K = base+{16..31}; packed 2 halves per VGPR. Each 8-run is
    // contiguous -> two float4 loads per source per run.
    v16h afrag[4];
#pragma unroll
    for (int s = 0; s < 4; ++s) {
        const int base = 32 * s + 16 * half;           // multiple of 16
        fill8(afrag[s], 0, ri4, rj4, zb4, base);       // K = base..base+7
        fill8(afrag[s], 8, ri4, rj4, zb4, base + 8);   // K = base+8..base+15
    }

    const int n0 = lane & 15;                   // this lane's output column (mod 16)
    float partial[8];
#pragma unroll
    for (int v = 0; v < 8; ++v) partial[v] = 0.0f;

    // ---- 4 N-tiles x 4 K-steps = 16 WMMAs per wave
#pragma unroll
    for (int t = 0; t < 4; ++t) {
        const int n = 16 * t + n0;
        const float bias2 = b2[n];
        v8f acc;
#pragma unroll
        for (int v = 0; v < 8; ++v) acc[v] = bias2;   // C starts at layer-2 bias

        // Transposed weights: lane's B-fragment (col n, K = kb..kb+15) is a
        // single contiguous 32-byte run -> two global_load_b128.
        const v16h* wrow = reinterpret_cast<const v16h*>(WhT + n * H1c);
#pragma unroll
        for (int s = 0; s < 4; ++s) {
            // B k order used here (contiguous, same as A): lanes 0-15 hold K = 32s+0..15,
            // lanes 16-31 hold K = 32s+16..31.
            const int kb = 32 * s + 16 * half;         // multiple of 16
            v16h bfrag = wrow[kb >> 4];
            acc = __builtin_amdgcn_wmma_f32_16x16x32_f16(
                /*neg_a=*/false, afrag[s], /*neg_b=*/false, bfrag,
                /*c_mod=*/(short)0, acc, /*reuse_a=*/false, /*reuse_b=*/false);
        }

        // layer 3 partial: relu(h2) * w3[n], per row v (M = v + 8*half)
        const float w3n = w3[n];
#pragma unroll
        for (int v = 0; v < 8; ++v) {
            float h2 = acc[v] > 0.0f ? acc[v] : 0.0f;
            partial[v] += h2 * w3n;
        }
    }

    // ---- reduce over the 16 columns held by this half-wave (xor butterfly
    // with masks 1,2,4,8 stays inside each 16-lane half)
#pragma unroll
    for (int off = 1; off < 16; off <<= 1) {
#pragma unroll
        for (int v = 0; v < 8; ++v)
            partial[v] += __shfl_xor(partial[v], off, 32);
    }

    const float bias3 = b3[0];
    if (n0 == 0) {                               // lanes 0 and 16 write 8 rows each
        const int rowbase = p0 + 8 * half;
        float* o = out + (size_t)zIdx * PAIRS;
#pragma unroll
        for (int v = 0; v < 8; ++v) {
            const int pp = rowbase + v;
            const int ii = pp / Nn, jj = pp % Nn;
            const float logit = partial[v] + bias3;
            float sg = 1.0f / (1.0f + __expf(-logit));
            if (ii == jj) sg = 0.0f;
            o[pp] = sg;
        }
    }
}

// ---------------------------------------------------------------------------
extern "C" void kernel_launch(void* const* d_in, const int* in_sizes, int n_in,
                              void* d_out, int out_size, void* d_ws, size_t ws_size,
                              hipStream_t stream) {
    // Inputs in setup_inputs() order:
    const float* ne      = (const float*)d_in[0];   // (B,N,D)
    const float* z_intra = (const float*)d_in[1];   // (B,ZI)
    const float* z_inter = (const float*)d_in[2];   // (B,ZE)
    const float* lag_emb = (const float*)d_in[3];   // (L,LE)
    const float* Wi1     = (const float*)d_in[4];   // (192,128)
    const float* bi1     = (const float*)d_in[5];   // (128)
    const float* Wi2     = (const float*)d_in[6];   // (128,64)
    const float* bi2     = (const float*)d_in[7];   // (64)
    const float* Wi3     = (const float*)d_in[8];   // (64,1)
    const float* bi3     = (const float*)d_in[9];   // (1)
    const float* Wl1     = (const float*)d_in[10];  // (200,128)
    const float* bl1     = (const float*)d_in[11];  // (128)
    const float* Wl2     = (const float*)d_in[12];  // (128,64)
    const float* bl2     = (const float*)d_in[13];  // (64)
    const float* Wl3     = (const float*)d_in[14];  // (64,1)
    const float* bl3     = (const float*)d_in[15];  // (1)

    float* W_out = (float*)d_out;                       // (B,N,N)
    float* A_out = (float*)d_out + Bz * PAIRS;          // (B,L,N,N)

    // Workspace layout (bytes)
    char* ws = (char*)d_ws;
    const size_t rowBytes = (size_t)Bz * Nn * H1c * sizeof(float);   // 327680
    float*    pi    = (float*)(ws);
    float*    pj    = (float*)(ws + rowBytes);
    float*    qi    = (float*)(ws + 2 * rowBytes);
    float*    qj    = (float*)(ws + 3 * rowBytes);
    float*    pzb   = (float*)(ws + 4 * rowBytes);                   // 512 f
    float*    qzl   = (float*)(ws + 4 * rowBytes + 512 * 4);         // 2048 f
    _Float16* Wi2hT = (_Float16*)(ws + 4 * rowBytes + 2560 * 4);     // 8192 h
    _Float16* Wl2hT = (_Float16*)(ws + 4 * rowBytes + 2560 * 4 + 8192 * 2);

    // Stage 1: row features (4 * 81920 = 327680 threads -> 1280 blocks)
    prep_rows_kernel<<<1280, 256, 0, stream>>>(ne, Wi1, Wl1, pi, pj, qi, qj);

    // Stage 2: z/lag bases + transposed f16 weights (18944 threads -> 74 blocks)
    prep_misc_kernel<<<74, 256, 0, stream>>>(z_intra, z_inter, lag_emb,
                                             Wi1, bi1, Wl1, bl1, Wi2, Wl2,
                                             pzb, qzl, Wi2hT, Wl2hT);

    // Stage 3a: W pass — Bz * 1600 groups = 6400 waves -> 800 blocks of 8 waves
    pair_mlp_wmma_kernel<<<(Bz * GROUPS_PER) / 8, 256, 0, stream>>>(
        pi, pj, pzb, Wi2hT, bi2, Wi3, bi3, W_out, /*Lg=*/1);

    // Stage 3b: A pass — Bz*L * 1600 groups = 25600 waves -> 3200 blocks
    pair_mlp_wmma_kernel<<<(Bz * Ll * GROUPS_PER) / 8, 256, 0, stream>>>(
        qi, qj, qzl, Wl2hT, bl2, Wl3, bl3, A_out, /*Lg=*/Ll);
}
